// GAPnet_35038343200904
// MI455X (gfx1250) — matrix operand order repaired and checked by reading, not compiled, not run on hardware
//
#include <hip/hip_runtime.h>
#include <hip/hip_bf16.h>

typedef __attribute__((ext_vector_type(16))) _Float16 v16h;
typedef __attribute__((ext_vector_type(8)))  float    v8f;

#define N_PTS   65536
#define NEG_INF (-3.0e38f)

// ---------------------------------------------------------------------------
// k_prep1: collapse the two affine conv stacks (single block, tiny work)
//   A3   (1024x3)  = t_m3_w @ t_m2_w @ t_m1_w          C3 (1024) combined bias
//   M321 (128x67)  = m3_w @ m2_w @ m1_w                BM (128)  combined bias
// ---------------------------------------------------------------------------
__global__ __launch_bounds__(256) void k_prep1(
    const float* __restrict__ tm1w, const float* __restrict__ tm1b,
    const float* __restrict__ tm2w, const float* __restrict__ tm2b,
    const float* __restrict__ tm3w, const float* __restrict__ tm3b,
    const float* __restrict__ m1w,  const float* __restrict__ m1b,
    const float* __restrict__ m2w,  const float* __restrict__ m2b,
    const float* __restrict__ m3w,  const float* __restrict__ m3b,
    float* __restrict__ A3, float* __restrict__ C3,
    float* __restrict__ M321, float* __restrict__ BM)
{
    __shared__ float sT1[128 * 3];
    __shared__ float sb12[128];
    __shared__ float sM21[128 * 67];
    __shared__ float sbm2[128];
    const int tid = threadIdx.x;

    // stage A: 128-row intermediates
    for (int idx = tid; idx < 128 * 3; idx += 256) {
        int r = idx / 3, d = idx % 3;
        float acc = 0.f;
        for (int c = 0; c < 64; ++c) acc += tm2w[r * 64 + c] * tm1w[c * 3 + d];
        sT1[idx] = acc;
    }
    for (int r = tid; r < 128; r += 256) {
        float a = tm2b[r], b = m2b[r];
        for (int c = 0; c < 64; ++c) {
            a += tm2w[r * 64 + c] * tm1b[c];
            b += m2w[r * 64 + c] * m1b[c];
        }
        sb12[r] = a;
        sbm2[r] = b;
    }
    for (int idx = tid; idx < 128 * 67; idx += 256) {
        int r = idx / 67, j = idx % 67;
        float acc = 0.f;
        for (int c = 0; c < 64; ++c) acc += m2w[r * 64 + c] * m1w[c * 67 + j];
        sM21[idx] = acc;
    }
    __syncthreads();

    // stage B: apply third layer of each stack
    for (int idx = tid; idx < 1024 * 3; idx += 256) {
        int o = idx / 3, d = idx % 3;
        float acc = 0.f;
        for (int c = 0; c < 128; ++c) acc += tm3w[o * 128 + c] * sT1[c * 3 + d];
        A3[idx] = acc;
    }
    for (int o = tid; o < 1024; o += 256) {
        float acc = tm3b[o];
        for (int c = 0; c < 128; ++c) acc += tm3w[o * 128 + c] * sb12[c];
        C3[o] = acc;
    }
    for (int idx = tid; idx < 128 * 67; idx += 256) {
        int r = idx / 67, j = idx % 67;
        float acc = 0.f;
        for (int c = 0; c < 128; ++c) acc += m3w[r * 128 + c] * sM21[c * 67 + j];
        M321[idx] = acc;
    }
    for (int r = tid; r < 128; r += 256) {
        float acc = m3b[r];
        for (int c = 0; c < 128; ++c) acc += m3w[r * 128 + c] * sbm2[c];
        BM[r] = acc;
    }
}

// ---------------------------------------------------------------------------
// k_wfrag: build the collapsed trunk weight matrix W (1024 x 160) as f16,
// stored directly in WMMA A-fragment order, contiguous k per lane half (same k order as B):
//   lanes 0-15:  M=lane,    elem e -> K = e
//   lanes 16-31: M=lane-16, elem e -> K = e+16
// columns: [0..66]=A67 (m4_w[:, :128] @ M321), [67..130]=m4_w[:,128:192],
//          [131]=combined bias (input col forced to 1.0), [132..159]=0 pad.
// ---------------------------------------------------------------------------
__global__ __launch_bounds__(256) void k_wfrag(
    const float* __restrict__ m4w, const float* __restrict__ m4b,
    const float* __restrict__ M321, const float* __restrict__ BM,
    _Float16* __restrict__ wfrag)
{
    __shared__ float sM[128 * 67];
    __shared__ float sbm[128];
    const int tid = threadIdx.x;
    const int mtb = blockIdx.x;      // M-tile index 0..63 (rows mtb*16..+15)

    for (int idx = tid; idx < 128 * 67; idx += 256) sM[idx] = M321[idx];
    for (int r = tid; r < 128; r += 256) sbm[r] = BM[r];
    __syncthreads();

    for (int idx = tid; idx < 16 * 160; idx += 256) {
        int m = idx / 160, j = idx % 160;
        int o = mtb * 16 + m;
        float val;
        if (j < 67) {
            val = 0.f;
            for (int c = 0; c < 128; ++c) val += m4w[o * 192 + c] * sM[c * 67 + j];
        } else if (j < 131) {
            val = m4w[o * 192 + 128 + (j - 67)];
        } else if (j == 131) {
            val = m4b[o];
            for (int c = 0; c < 128; ++c) val += m4w[o * 192 + c] * sbm[c];
        } else {
            val = 0.f;
        }
        int kt = j >> 5, kk = j & 31;
        int h = kk >> 4;
        int e = kk & 15;
        int lanep = h * 16 + m;
        wfrag[((((kt * 64 + mtb) * 32) + lanep) << 4) + e] = (_Float16)val;
    }
}

// ---------------------------------------------------------------------------
// k_tnetmax: hmax_partial[chunk][c] = max over 1024 pts of (A3[c]·x[:,n])
// ---------------------------------------------------------------------------
__global__ __launch_bounds__(256) void k_tnetmax(
    const float* __restrict__ x, const float* __restrict__ A3,
    const float* __restrict__ C3, float* __restrict__ hpart)
{
    __shared__ float xs[3 * 1024];
    const int tid = threadIdx.x;
    const int base = blockIdx.x * 1024;
    for (int idx = tid; idx < 3 * 1024; idx += 256) {
        int d = idx >> 10, pp = idx & 1023;
        xs[idx] = x[d * N_PTS + base + pp];
    }
    __syncthreads();
    for (int q = 0; q < 4; ++q) {
        int ch = tid + q * 256;
        float a0 = A3[ch * 3 + 0], a1 = A3[ch * 3 + 1], a2 = A3[ch * 3 + 2];
        float mx = NEG_INF;
        for (int pp = 0; pp < 1024; ++pp) {
            float v = a0 * xs[pp] + a1 * xs[1024 + pp] + a2 * xs[2048 + pp];
            mx = fmaxf(mx, v);
        }
        hpart[blockIdx.x * 1024 + ch] = mx + C3[ch];   // +bias commutes with max
    }
}

// ---------------------------------------------------------------------------
// k_tnetfc: reduce hmax partials, run t_f1/t_f2/t_f3 matvecs -> t9 (3x3 flat)
// ---------------------------------------------------------------------------
__global__ __launch_bounds__(256) void k_tnetfc(
    const float* __restrict__ hpart,
    const float* __restrict__ f1w, const float* __restrict__ f1b,
    const float* __restrict__ f2w, const float* __restrict__ f2b,
    const float* __restrict__ f3w, const float* __restrict__ f3b,
    float* __restrict__ t9)
{
    __shared__ float hm[1024];
    __shared__ float h1[512];
    __shared__ float h2[256];
    const int tid = threadIdx.x;
    for (int ch = tid; ch < 1024; ch += 256) {
        float mx = NEG_INF;
        for (int c = 0; c < 64; ++c) mx = fmaxf(mx, hpart[c * 1024 + ch]);
        hm[ch] = mx;
    }
    __syncthreads();
    for (int r = tid; r < 512; r += 256) {
        float acc = f1b[r];
        for (int k = 0; k < 1024; ++k) acc += f1w[r * 1024 + k] * hm[k];
        h1[r] = acc;
    }
    __syncthreads();
    for (int r = tid; r < 256; r += 256) {
        float acc = f2b[r];
        for (int k = 0; k < 512; ++k) acc += f2w[r * 512 + k] * h1[k];
        h2[r] = acc;
    }
    __syncthreads();
    if (tid < 9) {
        float acc = f3b[tid];
        for (int k = 0; k < 256; ++k) acc += f3w[tid * 256 + k] * h2[k];
        t9[tid] = acc;
    }
}

// ---------------------------------------------------------------------------
// k_main: the WMMA kernel. 256 blocks x 256 threads (8 waves).
// Per 16-point tile:
//   phase1: 16 threads/point compute gF (16ch x 10k) into LDS (flat c*10+kk)
//   phase2: attn[f]=sum_k flat[k*16+f], gmax[f]=max_k flat[k*16+f];
//           build the 160-entry f16 input vector in LDS in B-fragment order:
//           [xn0..2 | attn[f] rep x4 (f*4+m) | gmax[f] rep x4 | 1.0 | 0 pad]
//   phase3: wave w owns output rows [w*128, w*128+128): 5 K-steps x 8 M-tiles
//           of v_wmma_f32_16x16x32_f16 (A-fragments are loop-invariant and get
//           hoisted into extended VGPRs by the compiler -> W read once/block).
// Point-max: elementwise v_max into register runmax[] per tile; one shuffle
// reduction per block at the very end, written straight to global partials.
// ---------------------------------------------------------------------------
__global__ __launch_bounds__(256) void k_main(
    const float* __restrict__ xknn, const float* __restrict__ x,
    const float* __restrict__ gwF, const float* __restrict__ gbF,
    const float* __restrict__ t9g, const _Float16* __restrict__ wfrag,
    float* __restrict__ xpart)
{
    __shared__ float lds_gf[16][160];
    __shared__ alignas(32) _Float16 lds_b[16][160];

    const int tid  = threadIdx.x;
    const int lane = tid & 31;
    const int wave = tid >> 5;
    const int pt   = tid >> 4;     // point-in-tile 0..15 (compute phases)
    const int s    = tid & 15;     // sub-thread / channel 0..15

    // per-thread constants
    const float w0 = gwF[s * 3 + 0], w1 = gwF[s * 3 + 1], w2 = gwF[s * 3 + 2];
    const float bF = gbF[s];
    float t9r[9];
    #pragma unroll
    for (int q = 0; q < 9; ++q) t9r[q] = t9g[q];

    // running max over all 16 tiles, kept in registers
    v8f runmax[8];
    #pragma unroll
    for (int t = 0; t < 8; ++t)
        #pragma unroll
        for (int v = 0; v < 8; ++v) runmax[t][v] = NEG_INF;

    for (int tile = 0; tile < 16; ++tile) {
        const int ptile = blockIdx.x * 16 + tile;
        const int i = ptile * 16 + pt;                 // global point index
        const float* xk = xknn + (size_t)i * 30;       // (3 x 10) for point i

        // ---- phase 1: gF into LDS (flat index c*10+kk) ----
        #pragma unroll
        for (int kk = 0; kk < 10; ++kk) {
            float v = bF + w0 * xk[kk] + w1 * xk[10 + kk] + w2 * xk[20 + kk];
            lds_gf[pt][s * 10 + kk] = v;
        }
        __syncthreads();

        // ---- phase 2: attn / gmax (torch reshape quirk: feat flat = k*16+f) ----
        float attn = 0.f, gmax = NEG_INF;
        #pragma unroll
        for (int k = 0; k < 10; ++k) {
            float v = lds_gf[pt][k * 16 + s];
            attn += v;
            gmax = fmaxf(gmax, v);
        }
        _Float16 ha = (_Float16)attn, hg = (_Float16)gmax;
        #pragma unroll
        for (int m = 0; m < 4; ++m) {
            lds_b[pt][3 + s * 4 + m]  = ha;   // head_attn, interleaved by head
            lds_b[pt][67 + s * 4 + m] = hg;   // gpool
        }
        if (s < 3) {   // xn with the (3,N)->(N,3) flat-reshape quirk
            int g = 3 * i + s;
            int c = g >> 16, pos = g & (N_PTS - 1);
            float xv = t9r[c * 3 + 0] * x[pos]
                     + t9r[c * 3 + 1] * x[N_PTS + pos]
                     + t9r[c * 3 + 2] * x[2 * N_PTS + pos];
            lds_b[pt][s] = (_Float16)xv;
        }
        if (s == 3) lds_b[pt][131] = (_Float16)1.0f;   // bias column
        lds_b[pt][132 + s] = (_Float16)0.0f;
        if (s < 12) lds_b[pt][148 + s] = (_Float16)0.0f;
        __syncthreads();

        // ---- phase 3: WMMA, D(1024x16) = W(1024x160) x B(160x16) ----
        v8f acc[8];
        #pragma unroll
        for (int t = 0; t < 8; ++t)
            #pragma unroll
            for (int v = 0; v < 8; ++v) acc[t][v] = 0.0f;

        const v16h* wv = (const v16h*)wfrag;
        const int bn = lane & 15, bh = lane >> 4;
        #pragma unroll
        for (int kt = 0; kt < 5; ++kt) {
            v16h bfrag = *(const v16h*)&lds_b[bn][kt * 32 + bh * 16];
            #pragma unroll
            for (int t = 0; t < 8; ++t) {
                int mt = wave * 8 + t;
                v16h afrag = wv[(kt * 64 + mt) * 32 + lane];
                acc[t] = __builtin_amdgcn_wmma_f32_16x16x32_f16(
                    false, afrag, false, bfrag, (short)0, acc[t], false, false);
            }
        }

        // ---- fold this tile into the register-resident running max ----
        #pragma unroll
        for (int t = 0; t < 8; ++t)
            #pragma unroll
            for (int v = 0; v < 8; ++v)
                runmax[t][v] = fmaxf(runmax[t][v], acc[t][v]);

        __syncthreads();   // lds_b/lds_gf are rewritten next tile
    }

    // one cross-lane max reduction per block (over the 16 points = lane group)
    #pragma unroll
    for (int t = 0; t < 8; ++t) {
        #pragma unroll
        for (int v = 0; v < 8; ++v) {
            float val = runmax[t][v];
            for (int off = 1; off < 16; off <<= 1)
                val = fmaxf(val, __shfl_xor(val, off, 32));
            if ((lane & 15) == 0) {   // lanes 0 (M=v) and 16 (M=v+8)
                int m = (wave * 8 + t) * 16 + v + ((lane >> 4) ? 8 : 0);
                xpart[blockIdx.x * 1024 + m] = val;
            }
        }
    }
}

// ---------------------------------------------------------------------------
// k_final: reduce block maxes -> xpool (1024), then f1..f5 matvecs -> out(40)
// ---------------------------------------------------------------------------
__global__ __launch_bounds__(256) void k_final(
    const float* __restrict__ xpart,
    const float* __restrict__ f1w, const float* __restrict__ f1b,
    const float* __restrict__ f2w, const float* __restrict__ f2b,
    const float* __restrict__ f3w, const float* __restrict__ f3b,
    const float* __restrict__ f4w, const float* __restrict__ f4b,
    const float* __restrict__ f5w, const float* __restrict__ f5b,
    float* __restrict__ out)
{
    __shared__ float xp[1024];
    __shared__ float o1[2048];
    __shared__ float o2[1024];
    __shared__ float o3[512];
    __shared__ float o4[256];
    const int tid = threadIdx.x;
    for (int ch = tid; ch < 1024; ch += 256) {
        float mx = NEG_INF;
        for (int b = 0; b < 256; ++b) mx = fmaxf(mx, xpart[b * 1024 + ch]);
        xp[ch] = mx;
    }
    __syncthreads();
    for (int r = tid; r < 2048; r += 256) {
        float acc = f1b[r];
        for (int k = 0; k < 1024; ++k) acc += f1w[r * 1024 + k] * xp[k];
        o1[r] = acc;
    }
    __syncthreads();
    for (int r = tid; r < 1024; r += 256) {
        float acc = f2b[r];
        for (int k = 0; k < 2048; ++k) acc += f2w[r * 2048 + k] * o1[k];
        o2[r] = acc;
    }
    __syncthreads();
    for (int r = tid; r < 512; r += 256) {
        float acc = f3b[r];
        for (int k = 0; k < 1024; ++k) acc += f3w[r * 1024 + k] * o2[k];
        o3[r] = acc;
    }
    __syncthreads();
    for (int r = tid; r < 256; r += 256) {
        float acc = f4b[r];
        for (int k = 0; k < 512; ++k) acc += f4w[r * 512 + k] * o3[k];
        o4[r] = acc;
    }
    __syncthreads();
    if (tid < 40) {
        float acc = f5b[tid];
        for (int k = 0; k < 256; ++k) acc += f5w[tid * 256 + k] * o4[k];
        out[tid] = acc;
    }
}

// ---------------------------------------------------------------------------
extern "C" void kernel_launch(void* const* d_in, const int* in_sizes, int n_in,
                              void* d_out, int out_size, void* d_ws, size_t ws_size,
                              hipStream_t stream) {
    (void)in_sizes; (void)n_in; (void)out_size; (void)ws_size;

    const float* xknn = (const float*)d_in[0];
    const float* x    = (const float*)d_in[1];
    const float* tm1w = (const float*)d_in[2];  const float* tm1b = (const float*)d_in[3];
    const float* tm2w = (const float*)d_in[4];  const float* tm2b = (const float*)d_in[5];
    const float* tm3w = (const float*)d_in[6];  const float* tm3b = (const float*)d_in[7];
    const float* tf1w = (const float*)d_in[8];  const float* tf1b = (const float*)d_in[9];
    const float* tf2w = (const float*)d_in[10]; const float* tf2b = (const float*)d_in[11];
    const float* tf3w = (const float*)d_in[12]; const float* tf3b = (const float*)d_in[13];
    const float* gwF  = (const float*)d_in[14]; const float* gbF  = (const float*)d_in[15];
    // d_in[16], d_in[17] (gap_w1/gap_b1) are dead code: softmax over size-1 axis == 1
    const float* m1w  = (const float*)d_in[18]; const float* m1b  = (const float*)d_in[19];
    const float* m2w  = (const float*)d_in[20]; const float* m2b  = (const float*)d_in[21];
    const float* m3w  = (const float*)d_in[22]; const float* m3b  = (const float*)d_in[23];
    const float* m4w  = (const float*)d_in[24]; const float* m4b  = (const float*)d_in[25];
    const float* f1w  = (const float*)d_in[26]; const float* f1b  = (const float*)d_in[27];
    const float* f2w  = (const float*)d_in[28]; const float* f2b  = (const float*)d_in[29];
    const float* f3w  = (const float*)d_in[30]; const float* f3b  = (const float*)d_in[31];
    const float* f4w  = (const float*)d_in[32]; const float* f4b  = (const float*)d_in[33];
    const float* f5w  = (const float*)d_in[34]; const float* f5b  = (const float*)d_in[35];

    char* ws = (char*)d_ws;
    float*    A3    = (float*)(ws + 0);         // 1024*3  f32
    float*    C3    = (float*)(ws + 16384);     // 1024    f32
    float*    M321  = (float*)(ws + 32768);     // 128*67  f32
    float*    BM    = (float*)(ws + 98304);     // 128     f32
    float*    T9    = (float*)(ws + 102400);    // 9       f32
    float*    HPART = (float*)(ws + 131072);    // 64*1024 f32
    _Float16* WFRAG = (_Float16*)(ws + 393216); // 1024*160 f16 (A-frag order)
    float*    XPART = (float*)(ws + 786432);    // 256*1024 f32

    k_prep1<<<1, 256, 0, stream>>>(tm1w, tm1b, tm2w, tm2b, tm3w, tm3b,
                                   m1w, m1b, m2w, m2b, m3w, m3b,
                                   A3, C3, M321, BM);
    k_wfrag<<<64, 256, 0, stream>>>(m4w, m4b, M321, BM, WFRAG);
    k_tnetmax<<<64, 256, 0, stream>>>(x, A3, C3, HPART);
    k_tnetfc<<<1, 256, 0, stream>>>(HPART, tf1w, tf1b, tf2w, tf2b, tf3w, tf3b, T9);
    k_main<<<256, 256, 0, stream>>>(xknn, x, gwF, gbF, T9, WFRAG, XPART);
    k_final<<<1, 256, 0, stream>>>(XPART, f1w, f1b, f2w, f2b, f3w, f3b,
                                   f4w, f4b, f5w, f5b, (float*)d_out);
}
